// Seq2SeqAttention_58222576664869
// MI455X (gfx1250) — matrix operand order repaired and checked by reading, not compiled, not run on hardware
//
#include <hip/hip_runtime.h>
#include <hip/hip_bf16.h>

// Seq2Seq LSTM + attention for MI455X (gfx1250).
// All GEMMs via v_wmma_f32_16x16x32_bf16 (bf16 in, fp32 accum).
// Cell state c kept fp32; hidden h stored bf16 (feeds next WMMA directly).

#define B_   256
#define T_   512
#define I_   64
#define H_   512
#define G4_  2048        // 4*H
#define HOR_ 30
#define NQ_  3
#define KE0  576         // 64 + 512
#define KE1  1024        // 512 + 512
#define KD0  1056        // 513 + 512 padded to mult of 32
#define KD1  1024        // 512 + 512

typedef __attribute__((ext_vector_type(16))) __bf16 v16bf;
typedef __attribute__((ext_vector_type(8)))  float  v8f;

__device__ __forceinline__ float bf2f(__bf16 b) {
    unsigned short u = __builtin_bit_cast(unsigned short, b);
    unsigned int   x = ((unsigned int)u) << 16;
    return __builtin_bit_cast(float, x);
}
__device__ __forceinline__ __bf16 f2bf(float f) {
    unsigned int x = __builtin_bit_cast(unsigned int, f);
    unsigned int r = x + 0x7FFFu + ((x >> 16) & 1u);
    unsigned short h = (unsigned short)(r >> 16);
    return __builtin_bit_cast(__bf16, h);
}
__device__ __forceinline__ float sigf(float x) { return 1.0f / (1.0f + __expf(-x)); }

// ---------------------------------------------------------------------------
// Setup kernels
// ---------------------------------------------------------------------------
__global__ void k_pack_w(const float* w0, int k0, const float* w1, int k1,
                         __bf16* dst, int ktot, int rows) {
    int idx = blockIdx.x * blockDim.x + threadIdx.x;
    if (idx >= rows * ktot) return;
    int r = idx / ktot, k = idx % ktot;
    float v = 0.0f;
    if (k < k0)           v = w0[(size_t)r * k0 + k];
    else if (k < k0 + k1) v = w1[(size_t)r * k1 + (k - k0)];
    dst[idx] = f2bf(v);
}

__global__ void k_add_bias(const float* a, const float* b, float* dst, int n) {
    int i = blockIdx.x * blockDim.x + threadIdx.x;
    if (i < n) dst[i] = a[i] + b[i];
}

__global__ void k_cvt_bf(const float* src, __bf16* dst, int n) {
    int i = blockIdx.x * blockDim.x + threadIdx.x;
    if (i < n) dst[i] = f2bf(src[i]);
}

__global__ void k_zero_bf(__bf16* p, int n) {
    int i = blockIdx.x * blockDim.x + threadIdx.x;
    if (i < n) p[i] = f2bf(0.0f);
}
__global__ void k_zero_f(float* p, int n) {
    int i = blockIdx.x * blockDim.x + threadIdx.x;
    if (i < n) p[i] = 0.0f;
}

__global__ void k_init_inp(const float* x, float* inp) {
    int b = blockIdx.x * blockDim.x + threadIdx.x;
    if (b < B_) inp[b] = x[((size_t)b * T_ + (T_ - 1)) * I_ + 0];
}

// ---------------------------------------------------------------------------
// Fused LSTM step: gates = [a0|a1] @ Wfused^T + bias, then cell update.
// One wave per 16x16 (m,j) tile with 4 gate accumulators (i,f,g,o).
// Grid: 64 blocks x 256 threads = 512 waves = (256/16)*(512/16) tiles.
// ---------------------------------------------------------------------------
__global__ __launch_bounds__(256)
void k_lstm_step(const __bf16* __restrict__ a0, int lda0, int k0,
                 const __bf16* __restrict__ a1, int lda1,
                 const __bf16* __restrict__ W, int ktot,
                 const float* __restrict__ bias,
                 const float* __restrict__ c_in,
                 float* __restrict__ c_out,
                 __bf16* __restrict__ h_out) {
    int wave = (blockIdx.x * blockDim.x + threadIdx.x) >> 5;
    int lane = threadIdx.x & 31;
    int mt = wave >> 5;                 // 16 M-tiles
    int jt = wave & 31;                 // 32 J-tiles
    int m0 = mt * 16, j0 = jt * 16;
    int lg = lane >> 4, lm = lane & 15;

    v8f acc0 = {}, acc1 = {}, acc2 = {}, acc3 = {};
    int mA = m0 + lm;
    int nj = j0 + lm;
    const __bf16* wr0 = W + (size_t)(0 * H_ + nj) * ktot;
    const __bf16* wr1 = W + (size_t)(1 * H_ + nj) * ktot;
    const __bf16* wr2 = W + (size_t)(2 * H_ + nj) * ktot;
    const __bf16* wr3 = W + (size_t)(3 * H_ + nj) * ktot;

    for (int kt = 0; kt < ktot; kt += 32) {
        // A fragment (16x32 bf16): two 8-elem contiguous runs per lane
        union { uint4 q[2]; v16bf v; } ua;
        int s0 = kt + 8 * lg;
        int s1 = s0 + 16;
        const __bf16* p0 = (s0 < k0) ? (a0 + (size_t)mA * lda0 + s0)
                                     : (a1 + (size_t)mA * lda1 + (s0 - k0));
        const __bf16* p1 = (s1 < k0) ? (a0 + (size_t)mA * lda0 + s1)
                                     : (a1 + (size_t)mA * lda1 + (s1 - k0));
        ua.q[0] = *(const uint4*)p0;
        ua.q[1] = *(const uint4*)p1;

        // B fragments (32x16 bf16): two 8-elem contiguous runs per lane (same K order as A)
        int kb = kt + 8 * lg;
        union { uint4 q[2]; v16bf v; } ub0, ub1, ub2, ub3;
        ub0.q[0] = *(const uint4*)(wr0 + kb); ub0.q[1] = *(const uint4*)(wr0 + kb + 16);
        ub1.q[0] = *(const uint4*)(wr1 + kb); ub1.q[1] = *(const uint4*)(wr1 + kb + 16);
        ub2.q[0] = *(const uint4*)(wr2 + kb); ub2.q[1] = *(const uint4*)(wr2 + kb + 16);
        ub3.q[0] = *(const uint4*)(wr3 + kb); ub3.q[1] = *(const uint4*)(wr3 + kb + 16);
        __builtin_prefetch(wr0 + kb + 32, 0, 1);   // global_prefetch_b8

        acc0 = __builtin_amdgcn_wmma_f32_16x16x32_bf16(false, ua.v, false, ub0.v, (short)0, acc0, false, false);
        acc1 = __builtin_amdgcn_wmma_f32_16x16x32_bf16(false, ua.v, false, ub1.v, (short)0, acc1, false, false);
        acc2 = __builtin_amdgcn_wmma_f32_16x16x32_bf16(false, ua.v, false, ub2.v, (short)0, acc2, false, false);
        acc3 = __builtin_amdgcn_wmma_f32_16x16x32_bf16(false, ua.v, false, ub3.v, (short)0, acc3, false, false);
    }

    union { v8f v; float f[8]; } r0, r1, r2, r3;
    r0.v = acc0; r1.v = acc1; r2.v = acc2; r3.v = acc3;
    int j = j0 + lm;
    for (int r = 0; r < 8; ++r) {
        int m = m0 + r + 8 * lg;        // C/D layout: VGPR r -> M=r (+8 for lanes 16-31)
        float gi = r0.f[r] + bias[j];
        float gf = r1.f[r] + bias[H_ + j];
        float gg = r2.f[r] + bias[2 * H_ + j];
        float go = r3.f[r] + bias[3 * H_ + j];
        float co = c_in[(size_t)m * H_ + j];
        float cn = sigf(gf) * co + sigf(gi) * tanhf(gg);
        c_out[(size_t)m * H_ + j] = cn;
        h_out[(size_t)m * H_ + j] = f2bf(sigf(go) * tanhf(cn));
    }
}

// ---------------------------------------------------------------------------
// Plain WMMA GEMM: out[M x N] = A(bf16, lda) @ W^T (W is [N][K] bf16)
// One wave per 16x16 tile.
// ---------------------------------------------------------------------------
template <bool OUT_BF16>
__global__ __launch_bounds__(256)
void k_gemm(const __bf16* __restrict__ A, int lda,
            const __bf16* __restrict__ W, int K, int Ntiles,
            void* __restrict__ out, int ldc) {
    int wave = (blockIdx.x * blockDim.x + threadIdx.x) >> 5;
    int lane = threadIdx.x & 31;
    int mt = wave / Ntiles, nt = wave % Ntiles;
    int m0 = mt * 16, n0 = nt * 16;
    int lg = lane >> 4, lm = lane & 15;

    v8f acc = {};
    int mA = m0 + lm;
    const __bf16* wr = W + (size_t)(n0 + lm) * K;
    for (int kt = 0; kt < K; kt += 32) {
        union { uint4 q[2]; v16bf v; } ua, ub;
        int s0 = kt + 8 * lg;
        ua.q[0] = *(const uint4*)(A + (size_t)mA * lda + s0);
        ua.q[1] = *(const uint4*)(A + (size_t)mA * lda + s0 + 16);
        int kb = kt + 8 * lg;
        ub.q[0] = *(const uint4*)(wr + kb);
        ub.q[1] = *(const uint4*)(wr + kb + 16);
        acc = __builtin_amdgcn_wmma_f32_16x16x32_bf16(false, ua.v, false, ub.v, (short)0, acc, false, false);
    }
    union { v8f v; float f[8]; } rr; rr.v = acc;
    int n = n0 + lm;
    for (int r = 0; r < 8; ++r) {
        int m = m0 + r + 8 * lg;
        if (OUT_BF16) ((__bf16*)out)[(size_t)m * ldc + n] = f2bf(rr.f[r]);
        else          ((float*)out)[(size_t)m * ldc + n] = rr.f[r];
    }
}

// ---------------------------------------------------------------------------
// Attention: energies[b,t] = sum_g tanh(eproj[t,b,g] + dproj[b,g]) * v[g]
// One wave per (b,t).
// ---------------------------------------------------------------------------
__global__ __launch_bounds__(256)
void k_energies(const __bf16* __restrict__ eproj, const float* __restrict__ dproj,
                const float* __restrict__ v, float* __restrict__ energ) {
    int wave = (blockIdx.x * blockDim.x + threadIdx.x) >> 5;
    int lane = threadIdx.x & 31;
    int b = wave >> 9;        // / T_
    int t = wave & (T_ - 1);
    const __bf16* ep = eproj + ((size_t)t * B_ + b) * H_;
    const float*  dp = dproj + (size_t)b * H_;
    float s = 0.0f;
    for (int g = lane; g < H_; g += 32)
        s += tanhf(bf2f(ep[g]) + dp[g]) * v[g];
    for (int m = 16; m; m >>= 1) s += __shfl_xor(s, m, 32);
    if (lane == 0) energ[(size_t)b * T_ + t] = s;
}

__global__ __launch_bounds__(256)
void k_softmax(const float* __restrict__ energ, float* __restrict__ attnw) {
    __shared__ float red[256];
    int b = blockIdx.x, tid = threadIdx.x;
    const float* e = energ + (size_t)b * T_;
    float a = e[tid], c = e[tid + 256];
    red[tid] = fmaxf(a, c);
    __syncthreads();
    for (int s = 128; s; s >>= 1) {
        if (tid < s) red[tid] = fmaxf(red[tid], red[tid + s]);
        __syncthreads();
    }
    float mx = red[0];
    __syncthreads();
    float ea = __expf(a - mx), ec = __expf(c - mx);
    red[tid] = ea + ec;
    __syncthreads();
    for (int s = 128; s; s >>= 1) {
        if (tid < s) red[tid] += red[tid + s];
        __syncthreads();
    }
    float inv = 1.0f / red[0];
    attnw[(size_t)b * T_ + tid]       = ea * inv;
    attnw[(size_t)b * T_ + tid + 256] = ec * inv;
}

__global__ __launch_bounds__(512)
void k_ctx(const float* __restrict__ attnw, const __bf16* __restrict__ hs1,
           float* __restrict__ ctx) {
    __shared__ float aw[T_];
    int b = blockIdx.x, h = threadIdx.x;
    aw[h] = attnw[(size_t)b * T_ + h];
    __syncthreads();
    float s = 0.0f;
    const __bf16* base = hs1 + (size_t)b * H_ + h;
    for (int t = 0; t < T_; ++t)
        s += aw[t] * bf2f(base[(size_t)t * B_ * H_]);
    ctx[(size_t)b * H_ + h] = s;
}

__global__ void k_pack_dec0(const float* __restrict__ inp, const float* __restrict__ ctx,
                            const __bf16* __restrict__ h, __bf16* __restrict__ xp) {
    int idx = blockIdx.x * blockDim.x + threadIdx.x;
    if (idx >= B_ * KD0) return;
    int b = idx / KD0, k = idx % KD0;
    if (k < 1)          xp[idx] = f2bf(inp[b]);
    else if (k < 513)   xp[idx] = f2bf(ctx[(size_t)b * H_ + (k - 1)]);
    else if (k < 1025)  xp[idx] = h[(size_t)b * H_ + (k - 513)];
    else                xp[idx] = f2bf(0.0f);
}

__global__ __launch_bounds__(128)
void k_out_proj(const __bf16* __restrict__ h1, const float* __restrict__ Wout,
                const float* __restrict__ bout, float* __restrict__ out,
                float* __restrict__ inp, int step) {
    __shared__ float red[NQ_][128];
    int b = blockIdx.x, tid = threadIdx.x;
    float s0 = 0, s1 = 0, s2 = 0;
    for (int hh = tid; hh < H_; hh += 128) {
        float v = bf2f(h1[(size_t)b * H_ + hh]);
        s0 += v * Wout[0 * H_ + hh];
        s1 += v * Wout[1 * H_ + hh];
        s2 += v * Wout[2 * H_ + hh];
    }
    red[0][tid] = s0; red[1][tid] = s1; red[2][tid] = s2;
    __syncthreads();
    for (int s = 64; s; s >>= 1) {
        if (tid < s) {
            red[0][tid] += red[0][tid + s];
            red[1][tid] += red[1][tid + s];
            red[2][tid] += red[2][tid + s];
        }
        __syncthreads();
    }
    if (tid == 0) {
        float o0 = red[0][0] + bout[0];
        float o1 = red[1][0] + bout[1];
        float o2 = red[2][0] + bout[2];
        size_t o = ((size_t)b * HOR_ + step) * NQ_;
        out[o] = o0; out[o + 1] = o1; out[o + 2] = o2;
        inp[b] = o1;                    // MEDIAN_IDX = 1
    }
}

// ---------------------------------------------------------------------------
// Host orchestration
// ---------------------------------------------------------------------------
static size_t aln(size_t x) { return (x + 255) & ~(size_t)255; }

extern "C" void kernel_launch(void* const* d_in, const int* in_sizes, int n_in,
                              void* d_out, int out_size, void* d_ws, size_t ws_size,
                              hipStream_t stream) {
    const float* x        = (const float*)d_in[0];
    const float* eWih0    = (const float*)d_in[1];
    const float* eWhh0    = (const float*)d_in[2];
    const float* ebih0    = (const float*)d_in[3];
    const float* ebhh0    = (const float*)d_in[4];
    const float* eWih1    = (const float*)d_in[5];
    const float* eWhh1    = (const float*)d_in[6];
    const float* ebih1    = (const float*)d_in[7];
    const float* ebhh1    = (const float*)d_in[8];
    const float* dWih0    = (const float*)d_in[9];
    const float* dWhh0    = (const float*)d_in[10];
    const float* dbih0    = (const float*)d_in[11];
    const float* dbhh0    = (const float*)d_in[12];
    const float* dWih1    = (const float*)d_in[13];
    const float* dWhh1    = (const float*)d_in[14];
    const float* dbih1    = (const float*)d_in[15];
    const float* dbhh1    = (const float*)d_in[16];
    const float* attnW    = (const float*)d_in[17];
    const float* attnv    = (const float*)d_in[18];
    const float* outW     = (const float*)d_in[19];
    const float* outb     = (const float*)d_in[20];
    float* out = (float*)d_out;

    // Workspace carve-up (~307 MB total)
    char* p = (char*)d_ws;
    auto take = [&](size_t bytes) { char* r = p; p += aln(bytes); return r; };
    __bf16* W0f   = (__bf16*)take((size_t)G4_ * KE0 * 2);
    __bf16* W1f   = (__bf16*)take((size_t)G4_ * KE1 * 2);
    __bf16* Wd0f  = (__bf16*)take((size_t)G4_ * KD0 * 2);
    __bf16* Wd1f  = (__bf16*)take((size_t)G4_ * KD1 * 2);
    __bf16* Waf   = (__bf16*)take((size_t)H_ * H_ * 2);
    float*  b0c   = (float*)take(G4_ * 4);
    float*  b1c   = (float*)take(G4_ * 4);
    float*  bd0c  = (float*)take(G4_ * 4);
    float*  bd1c  = (float*)take(G4_ * 4);
    __bf16* xbf   = (__bf16*)take((size_t)B_ * T_ * I_ * 2);
    __bf16* zbf   = (__bf16*)take((size_t)B_ * H_ * 2);
    __bf16* h0a   = (__bf16*)take((size_t)B_ * H_ * 2);
    __bf16* h0b   = (__bf16*)take((size_t)B_ * H_ * 2);
    __bf16* hdA   = (__bf16*)take((size_t)B_ * H_ * 2);
    __bf16* hdB   = (__bf16*)take((size_t)B_ * H_ * 2);
    __bf16* h0buf = (__bf16*)take((size_t)B_ * H_ * 2);
    float*  c0    = (float*)take((size_t)B_ * H_ * 4);
    float*  c1    = (float*)take((size_t)B_ * H_ * 4);
    float*  cscr  = (float*)take((size_t)B_ * H_ * 4);
    __bf16* hs1   = (__bf16*)take((size_t)T_ * B_ * H_ * 2);   // 134 MB
    __bf16* eproj = (__bf16*)take((size_t)T_ * B_ * H_ * 2);   // 134 MB
    float*  dproj = (float*)take((size_t)B_ * H_ * 4);
    float*  energ = (float*)take((size_t)B_ * T_ * 4);
    float*  attnw = (float*)take((size_t)B_ * T_ * 4);
    float*  ctx   = (float*)take((size_t)B_ * H_ * 4);
    __bf16* xp0   = (__bf16*)take((size_t)B_ * KD0 * 2);
    float*  inp   = (float*)take((size_t)B_ * 4);
    (void)ws_size; (void)n_in; (void)in_sizes; (void)out_size;

    // --- Setup: weights -> fused bf16, biases combined, x -> bf16, zeros ---
    k_pack_w<<<(G4_ * KE0 + 255) / 256, 256, 0, stream>>>(eWih0, I_, eWhh0, H_, W0f, KE0, G4_);
    k_pack_w<<<(G4_ * KE1 + 255) / 256, 256, 0, stream>>>(eWih1, H_, eWhh1, H_, W1f, KE1, G4_);
    k_pack_w<<<(G4_ * KD0 + 255) / 256, 256, 0, stream>>>(dWih0, 513, dWhh0, H_, Wd0f, KD0, G4_);
    k_pack_w<<<(G4_ * KD1 + 255) / 256, 256, 0, stream>>>(dWih1, H_, dWhh1, H_, Wd1f, KD1, G4_);
    k_pack_w<<<(H_ * H_ + 255) / 256, 256, 0, stream>>>(attnW, H_, (const float*)nullptr, 0, Waf, H_, H_);
    k_add_bias<<<(G4_ + 255) / 256, 256, 0, stream>>>(ebih0, ebhh0, b0c, G4_);
    k_add_bias<<<(G4_ + 255) / 256, 256, 0, stream>>>(ebih1, ebhh1, b1c, G4_);
    k_add_bias<<<(G4_ + 255) / 256, 256, 0, stream>>>(dbih0, dbhh0, bd0c, G4_);
    k_add_bias<<<(G4_ + 255) / 256, 256, 0, stream>>>(dbih1, dbhh1, bd1c, G4_);
    k_cvt_bf<<<(B_ * T_ * I_ + 255) / 256, 256, 0, stream>>>(x, xbf, B_ * T_ * I_);
    k_zero_bf<<<(B_ * H_ + 255) / 256, 256, 0, stream>>>(zbf, B_ * H_);
    k_zero_f<<<(B_ * H_ + 255) / 256, 256, 0, stream>>>(c0, B_ * H_);
    k_zero_f<<<(B_ * H_ + 255) / 256, 256, 0, stream>>>(c1, B_ * H_);
    k_init_inp<<<1, 256, 0, stream>>>(x, inp);

    // --- Encoder: 512 sequential fused WMMA LSTM steps per layer ---
    for (int t = 0; t < T_; ++t) {
        const __bf16* hp0 = (t == 0) ? zbf : ((t & 1) ? h0a : h0b);
        __bf16* hc0 = (t & 1) ? h0b : h0a;
        // layer 0: A = [x_t | h_prev], K = 64 + 512
        k_lstm_step<<<64, 256, 0, stream>>>(xbf + (size_t)t * I_, T_ * I_, I_,
                                            hp0, H_, W0f, KE0, b0c, c0, c0, hc0);
        // layer 1: A = [hs0_t | h1_prev], K = 512 + 512; h_out -> hs1[t]
        const __bf16* hp1 = (t == 0) ? zbf : (hs1 + (size_t)(t - 1) * B_ * H_);
        k_lstm_step<<<64, 256, 0, stream>>>(hc0, H_, H_,
                                            hp1, H_, W1f, KE1, b1c, c1, c1,
                                            hs1 + (size_t)t * B_ * H_);
    }

    // --- enc_proj: (T*B, 512) x (512, 512) WMMA GEMM, bf16 out ---
    {
        int waves = (T_ * B_ / 16) * (H_ / 16);   // 262144
        k_gemm<true><<<waves / 8, 256, 0, stream>>>(hs1, H_, Waf, H_, H_ / 16, eproj, H_);
    }

    // --- Decoder: 30 attention + 2-cell steps (carry h,c = layer-1 finals) ---
    for (int s = 0; s < HOR_; ++s) {
        const __bf16* hc = (s == 0) ? (hs1 + (size_t)(T_ - 1) * B_ * H_)
                                    : ((s & 1) ? hdB : hdA);
        __bf16* hn = (s & 1) ? hdA : hdB;
        // dec_proj = h @ attn_W^T (fp32 out)
        k_gemm<false><<<64, 256, 0, stream>>>(hc, H_, Waf, H_, H_ / 16, dproj, H_);
        k_energies<<<B_ * T_ / 8, 256, 0, stream>>>(eproj, dproj, attnv, energ);
        k_softmax<<<B_, 256, 0, stream>>>(energ, attnw);
        k_ctx<<<B_, 512, 0, stream>>>(attnw, hs1, ctx);
        k_pack_dec0<<<(B_ * KD0 + 255) / 256, 256, 0, stream>>>(inp, ctx, hc, xp0);
        // cell0: gates = [inp|ctx] @ Wih0^T + h @ Whh0^T  (c0 result unused)
        k_lstm_step<<<64, 256, 0, stream>>>(xp0, KD0, KD0, (const __bf16*)nullptr, 0,
                                            Wd0f, KD0, bd0c, c1, cscr, h0buf);
        // cell1: gates = h0 @ Wih1^T + h @ Whh1^T; carry (h1, c1)
        k_lstm_step<<<64, 256, 0, stream>>>(h0buf, H_, H_, hc, H_,
                                            Wd1f, KD1, bd1c, c1, c1, hn);
        k_out_proj<<<B_, 128, 0, stream>>>(hn, outW, outb, out, inp, s);
    }
}
